// Attention_32744830665026
// MI455X (gfx1250) — matrix operand order repaired and checked by reading, not compiled, not run on hardware
//
#include <hip/hip_runtime.h>

#define ALPHA 0.2f
#define CUT2 4.0f
#define NEG_BIG (-9.0e15f)

typedef __attribute__((ext_vector_type(16))) __bf16 v16bf;
typedef __attribute__((ext_vector_type(8)))  __bf16 v8bf;
typedef __attribute__((ext_vector_type(8)))  float  v8f;
typedef int v4i __attribute__((vector_size(16)));   // matches builtin's V4i

static constexpr int Bb = 8;
static constexpr int Nn = 2048;
static constexpr int Ff = 128;

#if __has_builtin(__builtin_amdgcn_global_load_async_to_lds_b128) && \
    __has_builtin(__builtin_amdgcn_s_wait_asynccnt)
#define USE_ASYNC_LDS 1
typedef __attribute__((address_space(1))) v4i* gv4i_p;
typedef __attribute__((address_space(3))) v4i* lv4i_p;
#endif

// ---------------------------------------------------------------------------
// Kernel 0: W (128x128 fp32, k-major) -> WT (128x128 bf16, f-major)
// ---------------------------------------------------------------------------
__global__ __launch_bounds__(256) void k_wt(const float* __restrict__ W,
                                            __bf16* __restrict__ WT) {
  int idx = blockIdx.x * 256 + threadIdx.x;   // 0..16383
  int k = idx >> 7;
  int f = idx & 127;
  WT[f * Ff + k] = (__bf16)W[idx];
}

// ---------------------------------------------------------------------------
// Kernel 1: Wh = x @ W via bf16 WMMA.  Emits:
//   WhT (b, f, j) bf16  — transposed so k_attn B tiles are contiguous
//   f1  (b, j) fp32
//   jd  (b, j) float4 = {f2, px, py, pz} — packed so k_attn can bulk-stage it
// One wave per 16-row i-tile (all 128 f columns).
// ---------------------------------------------------------------------------
__global__ __launch_bounds__(256) void k_wh(const float*  __restrict__ x,
                                            const __bf16* __restrict__ WT,
                                            const float*  __restrict__ a,
                                            const float*  __restrict__ pos,
                                            __bf16* __restrict__ WhT,
                                            float*  __restrict__ f1,
                                            float4* __restrict__ jd) {
  const int lane = threadIdx.x & 31;
  const int wave = threadIdx.x >> 5;
  const int t    = blockIdx.x * 8 + wave;     // 0..1023 tile id
  const int b    = t >> 7;                    // 128 tiles per batch
  const int i0   = (t & 127) << 4;
  const int l16  = lane & 15;
  const int hi   = lane >> 4;                 // 0 / 1 half

  v8f acc[8] = {};
  const int row = i0 + l16;
  const float* xrow = x + (size_t)(b * Nn + row) * Ff;

#pragma unroll
  for (int kc = 0; kc < 4; ++kc) {
    const int kbase = kc * 32 + hi * 16;
    v8f xlo = *(const v8f*)(xrow + kbase);
    v8f xhi = *(const v8f*)(xrow + kbase + 8);
    // batch all 8 B tiles so loads overlap, WMMAs issue back-to-back
    const int koff = kc * 32 + hi * 16;
    v16bf Bt[8];
#pragma unroll
    for (int c = 0; c < 8; ++c)
      Bt[c] = *(const v16bf*)(WT + (c * 16 + l16) * Ff + koff);
    v16bf A;
#pragma unroll
    for (int e = 0; e < 8; ++e) {
      A[e]     = (__bf16)xlo[e];
      A[e + 8] = (__bf16)xhi[e];
    }
#pragma unroll
    for (int c = 0; c < 8; ++c)
      acc[c] = __builtin_amdgcn_wmma_f32_16x16x32_bf16(
          false, A, false, Bt[c], (short)0, acc[c], false, false);
  }

  // Store WhT bf16: C element (M=r+8*hi, N=c*16+l16) -> WhT[b][f][i]
#pragma unroll
  for (int c = 0; c < 8; ++c) {
    v8bf hv;
#pragma unroll
    for (int r = 0; r < 8; ++r) hv[r] = (__bf16)acc[c][r];
    __bf16* dst = WhT + (size_t)(b * Ff + c * 16 + l16) * Nn + i0 + hi * 8;
    *(v8bf*)dst = hv;
  }

  // f1/f2: row dot with a1/a2, reduce across each 16-lane half
  float p1[8], p2[8];
#pragma unroll
  for (int r = 0; r < 8; ++r) { p1[r] = 0.f; p2[r] = 0.f; }
#pragma unroll
  for (int c = 0; c < 8; ++c) {
    const float a1 = a[c * 16 + l16];
    const float a2 = a[128 + c * 16 + l16];
#pragma unroll
    for (int r = 0; r < 8; ++r) {
      p1[r] += acc[c][r] * a1;
      p2[r] += acc[c][r] * a2;
    }
  }
#pragma unroll
  for (int off = 1; off < 16; off <<= 1) {
#pragma unroll
    for (int r = 0; r < 8; ++r) {
      p1[r] += __shfl_xor(p1[r], off, 32);
      p2[r] += __shfl_xor(p2[r], off, 32);
    }
  }
  if (l16 == 0) {
#pragma unroll
    for (int r = 0; r < 8; ++r) {
      const int ri = i0 + hi * 8 + r;
      f1[b * Nn + ri] = p1[r];
      float4 v;
      v.x = p2[r];
      v.y = pos[(size_t)(b * Nn + ri) * 3 + 0];
      v.z = pos[(size_t)(b * Nn + ri) * 3 + 1];
      v.w = pos[(size_t)(b * Nn + ri) * 3 + 2];
      jd[b * Nn + ri] = v;
    }
  }
}

// ---------------------------------------------------------------------------
// Kernel 2: fused masked softmax-attention, flash style.
// Per block (one batch): stage all per-j data {f2,px,py,pz} in LDS (32 KB,
// async-to-LDS when available).  One wave per 16-row i-tile; stream j in
// chunks of 32; fp32 online softmax with conditional rescale; P@Wh via
// 8x v_wmma_f32_16x16x32_bf16 per chunk with batched B-tile loads; ELU.
// ---------------------------------------------------------------------------
__global__ __launch_bounds__(256) void k_attn(const float*  __restrict__ f1g,
                                              const float4* __restrict__ jd,
                                              const __bf16* __restrict__ WhT,
                                              float* __restrict__ out) {
  __shared__ float4 jdata[Nn];                      // 32 KB

  const int lane = threadIdx.x & 31;
  const int wave = threadIdx.x >> 5;
  const int t    = blockIdx.x * 8 + wave;
  const int b    = t >> 7;                          // uniform across block
  const int i0   = (t & 127) << 4;
  const int l16  = lane & 15;
  const int hi   = lane >> 4;

  // ---- stage per-j data for the whole batch row range into LDS ----
  const float4* jdg = jd + (size_t)b * Nn;
#ifdef USE_ASYNC_LDS
  for (int jj = threadIdx.x; jj < Nn; jj += 256) {
    __builtin_amdgcn_global_load_async_to_lds_b128(
        (gv4i_p)(jdg + jj), (lv4i_p)&jdata[jj], 0, 0);
  }
  __builtin_amdgcn_s_wait_asynccnt(0);
#else
  for (int jj = threadIdx.x; jj < Nn; jj += 256) jdata[jj] = jdg[jj];
#endif
  __syncthreads();

  const int   row = i0 + l16;
  const float f1v = f1g[b * Nn + row];
  const float4 pi = jdata[row];                     // .y/.z/.w = position_i

  v8f acc[8] = {};
  float m = -3.0e38f;
  float l = 0.0f;

  const __bf16* whb = WhT + (size_t)b * Ff * Nn;
  const int kbase = hi * 16;

  for (int j0 = 0; j0 < Nn; j0 += 32) {
    // issue all 8 B-tile loads first; score VALU work hides their latency
    const int joff = j0 + hi * 16;
    v16bf Bt[8];
#pragma unroll
    for (int c = 0; c < 8; ++c)
      Bt[c] = *(const v16bf*)(whb + (size_t)(c * 16 + l16) * Nn + joff);

    float s[16];
    float cmax = -3.0e38f;
#pragma unroll
    for (int e = 0; e < 16; ++e) {
      const int k = kbase + e;                      // A-matrix K index
      const float4 jv = jdata[j0 + k];              // LDS broadcast read
      const float dx = pi.y - jv.y;
      const float dy = pi.z - jv.z;
      const float dz = pi.w - jv.w;
      const float d2 = dx * dx + dy * dy + dz * dz;
      const float tv = f1v + jv.x;
      float sv = tv > 0.f ? tv : ALPHA * tv;        // leaky_relu
      sv = d2 > CUT2 ? NEG_BIG : sv;                // distance mask
      s[e] = sv;
      cmax = fmaxf(cmax, sv);
    }
    cmax = fmaxf(cmax, __shfl_xor(cmax, 16, 32));   // row max across halves
    const float mnew = fmaxf(m, cmax);

    // conditional rescale: skip once the running max has settled
    if (__ballot(mnew > m) != 0ull) {
      const float corr = __expf(m - mnew);
      l *= corr;
      float rs[8];
#pragma unroll
      for (int r = 0; r < 8; ++r) rs[r] = __shfl(corr, r + hi * 8, 32);
#pragma unroll
      for (int c = 0; c < 8; ++c)
#pragma unroll
        for (int r = 0; r < 8; ++r) acc[c][r] *= rs[r];
    }
    m = mnew;

    float psum = 0.f;
    v16bf A;
#pragma unroll
    for (int e = 0; e < 16; ++e) {
      const float p = __expf(s[e] - m);
      psum += p;
      A[e] = (__bf16)p;
    }
    psum += __shfl_xor(psum, 16, 32);
    l += psum;

    // P(16x32) @ Wh(32x128): 8 WMMAs off the pre-loaded B tiles
#pragma unroll
    for (int c = 0; c < 8; ++c)
      acc[c] = __builtin_amdgcn_wmma_f32_16x16x32_bf16(
          false, A, false, Bt[c], (short)0, acc[c], false, false);
  }

  // normalize + ELU + store (coalesced b32 stores across each half-wave)
  const float invl = 1.0f / l;
  float is[8];
#pragma unroll
  for (int r = 0; r < 8; ++r) is[r] = __shfl(invl, r + hi * 8, 32);
#pragma unroll
  for (int c = 0; c < 8; ++c) {
#pragma unroll
    for (int r = 0; r < 8; ++r) {
      float v = acc[c][r] * is[r];
      v = v > 0.f ? v : (__expf(v) - 1.0f);         // elu
      out[(size_t)(b * Nn + i0 + hi * 8 + r) * Ff + c * 16 + l16] = v;
    }
  }
}

// ---------------------------------------------------------------------------
extern "C" void kernel_launch(void* const* d_in, const int* in_sizes, int n_in,
                              void* d_out, int out_size, void* d_ws, size_t ws_size,
                              hipStream_t stream) {
  (void)in_sizes; (void)n_in; (void)out_size; (void)ws_size;
  const float* x        = (const float*)d_in[0];
  const float* position = (const float*)d_in[1];
  const float* W        = (const float*)d_in[2];
  const float* a        = (const float*)d_in[3];
  float* out = (float*)d_out;

  char* ws = (char*)d_ws;
  __bf16* WT  = (__bf16*)ws;                                   // 32 KB
  __bf16* WhT = (__bf16*)(ws + 32768);                         // 4 MB
  float*  f1  = (float*)(ws + 32768 + (size_t)Bb * Ff * Nn * 2);
  float4* jdp = (float4*)(ws + 32768 + (size_t)Bb * Ff * Nn * 2
                               + (size_t)Bb * Nn * 4);         // 256 KB

  k_wt  <<<64, 256, 0, stream>>>(W, WT);
  k_wh  <<<128, 256, 0, stream>>>(x, WT, a, position, WhT, f1, jdp);
  k_attn<<<128, 256, 0, stream>>>(f1, jdp, WhT, out);
}
